// Self_Attn_292057776546
// MI455X (gfx1250) — matrix operand order repaired and checked by reading, not compiled, not run on hardware
//
#include <hip/hip_runtime.h>

// ---------- types ----------
typedef __attribute__((ext_vector_type(16))) __bf16        v16bf;
typedef __attribute__((ext_vector_type(8)))  float         v8f;
typedef __attribute__((ext_vector_type(4)))  unsigned int  u32x4;
typedef __attribute__((ext_vector_type(2)))  unsigned int  u32x2;
typedef __attribute__((ext_vector_type(4)))  float         f32x4;
typedef unsigned short u16;
typedef int v4i __attribute__((__vector_size__(16)));   // matches builtin param type

union FragBF { v16bf v; u32x4 u[2]; };

__device__ __forceinline__ u16 f2bf(float f) {
  unsigned int u = __float_as_uint(f);
  u += 0x7fffu + ((u >> 16) & 1u);       // round-to-nearest-even
  return (u16)(u >> 16);
}

// ---------- CDNA5 async global->LDS path (probe via __has_builtin) ----------
#if defined(__HIP_DEVICE_COMPILE__) && __has_builtin(__builtin_amdgcn_global_load_async_to_lds_b128)
#define HAVE_ASYNC_LDS 1
#else
#define HAVE_ASYNC_LDS 0
#endif

__device__ __forceinline__ void async_copy16(u16* ldst, const u16* gsrc) {
#if HAVE_ASYNC_LDS
  __builtin_amdgcn_global_load_async_to_lds_b128(
      (__attribute__((address_space(1))) v4i*)(u16*)gsrc,
      (__attribute__((address_space(3))) v4i*)ldst, 0, 0);
#else
  *(u32x4*)ldst = *(const u32x4*)gsrc;   // sync fallback (global_load + ds_store)
#endif
}

template <int N>
__device__ __forceinline__ void wait_async() {
#if HAVE_ASYNC_LDS
#if __has_builtin(__builtin_amdgcn_s_wait_asynccnt)
  __builtin_amdgcn_s_wait_asynccnt((unsigned short)N);
#else
  asm volatile("s_wait_asynccnt %0" :: "i"(N) : "memory");
#endif
#endif
}

// ---------- problem constants (fixed by the reference) ----------
#define BATCH 8
#define CCH   64         // C
#define WH    65536      // 256*256 pixels
#define NTOK  1024       // (256/8)^2 tokens
#define MTOK  512        // Cq*P*P = 8*64
#define KSTEP 32

// =====================================================================
// Kernel 1: fused q/k/v 1x1-conv + patch-token scatter, fp32 -> bf16.
//   qT,kT stored [B][N=1024][M=512]; v stored [B][M=512][N=1024].
// =====================================================================
__global__ void __launch_bounds__(256) qkv_patch_kernel(
    const float* __restrict__ x,
    const float* __restrict__ Wq, const float* __restrict__ bq,
    const float* __restrict__ Wk, const float* __restrict__ bk,
    const float* __restrict__ Wv, const float* __restrict__ bv,
    u16* __restrict__ qT, u16* __restrict__ kT, u16* __restrict__ vTok)
{
  __shared__ float sWq[512], sWk[512], sWv[512];   // transposed [c][cq]
  int t = threadIdx.x;
  for (int i = t; i < 512; i += 256) {
    int cq = i >> 6, c = i & 63;
    sWq[c * 8 + cq] = Wq[i];
    sWk[c * 8 + cq] = Wk[i];
    sWv[c * 8 + cq] = Wv[i];
  }
  __syncthreads();

  int gid = blockIdx.x * 256 + t;
  int b   = gid >> 16;             // 65536 pixels per batch
  int pix = gid & (WH - 1);
  const float* xp = x + (size_t)b * CCH * WH + pix;

  float q[8], k[8], v[8];
#pragma unroll
  for (int cq = 0; cq < 8; ++cq) { q[cq] = bq[cq]; k[cq] = bk[cq]; v[cq] = bv[cq]; }

  for (int c = 0; c < 64; ++c) {
    float xv = xp[(size_t)c * WH];
#pragma unroll
    for (int cq = 0; cq < 8; ++cq) {
      q[cq] = fmaf(sWq[c * 8 + cq], xv, q[cq]);
      k[cq] = fmaf(sWk[c * 8 + cq], xv, k[cq]);
      v[cq] = fmaf(sWv[c * 8 + cq], xv, v[cq]);
    }
  }

  // replicate reference unfold->view index permutation
  int w = pix >> 8, h = pix & 255;
  int wp = w >> 3, pw = w & 7, hp = h >> 3, ph = h & 7;
  int n   = (hp & 15) * 64 + pw * 8 + ph;     // token column, cq-independent
  int mlo = wp * 2 + (hp >> 4);               // m = cq*64 + mlo

  u16* qrow = qT + ((size_t)b * NTOK + n) * MTOK;
  u16* krow = kT + ((size_t)b * NTOK + n) * MTOK;
  u16* vb   = vTok + (size_t)b * MTOK * NTOK;
#pragma unroll
  for (int cq = 0; cq < 8; ++cq) {
    int m = cq * 64 + mlo;
    qrow[m] = f2bf(q[cq]);
    krow[m] = f2bf(k[cq]);
    vb[(size_t)m * NTOK + n] = f2bf(v[cq]);
  }
}

// =====================================================================
// Kernel 2: NT bf16 GEMM via v_wmma_f32_16x16x32_bf16.
//   C[b][i][j] = sum_k A[b][i][k] * B[b][j][k]   (row-major, f32 out)
//   Block = 8 waves, 256x64 output tile. B tile (64 rows x 32 K) staged
//   in LDS via async global->LDS copies, double-buffered (ASYNCcnt).
//   Each wave: 32x64 strip = 2 A-frags x 4 B-frags -> 8 WMMAs / K-step.
// =====================================================================
__global__ void __launch_bounds__(256) gemm_nt_bf16_wmma(
    const u16* __restrict__ A, const u16* __restrict__ B, float* __restrict__ C,
    int K, int Nout, int rowBlocks, int nStrips,
    unsigned long long sA, unsigned long long sB, unsigned long long sC)
{
  __shared__ u16 sBt[2][64 * KSTEP];            // 2 x 4 KiB double buffer

  int blocksPerBatch = rowBlocks * nStrips;
  int b  = blockIdx.x / blocksPerBatch;
  int r  = blockIdx.x - b * blocksPerBatch;
  int mblk  = (r / nStrips) * 256;
  int nbase = (r % nStrips) * 64;

  int lane = threadIdx.x;
  int wv   = threadIdx.y;
  int t    = wv * 32 + lane;
  int l15  = lane & 15;
  int g    = lane >> 4;                          // lane half-group

  const u16* Ab = A + (size_t)b * sA;
  const u16* Bb = B + (size_t)b * sB;
  float*     Cb = C + (size_t)b * sC;

  // cooperative B-tile staging: thread t copies 16B (row t>>2, chunk t&3)
  int crow = t >> 2;
  int cchk = (t & 3) * 8;                        // halves
  const u16* gB = Bb + (size_t)(nbase + crow) * K + cchk;
  u16* lBp[2] = { &sBt[0][crow * KSTEP + cchk], &sBt[1][crow * KSTEP + cchk] };

  // wave tile: rows [mbase, mbase+32)
  int mbase = mblk + wv * 32;
  size_t aOff0 = (size_t)(mbase + l15) * K;
  size_t aOff1 = (size_t)(mbase + 16 + l15) * K;

  v8f acc[2][4] = {};

  async_copy16(lBp[0], gB);                      // prologue: stage k0 = 0

  for (int k0 = 0; k0 < K; k0 += KSTEP) {
    int cur = (k0 / KSTEP) & 1;
    if (k0 + KSTEP < K) {
      async_copy16(lBp[cur ^ 1], gB + k0 + KSTEP);
      wait_async<1>();                           // current buffer's op done
    } else {
      wait_async<0>();
    }
    __syncthreads();                             // tile visible to all waves

    FragBF fa0, fa1;
    fa0.u[0] = *(const u32x4*)(Ab + aOff0 + k0 + g * 8);
    fa0.u[1] = *(const u32x4*)(Ab + aOff0 + k0 + 16 + g * 8);
    fa1.u[0] = *(const u32x4*)(Ab + aOff1 + k0 + g * 8);
    fa1.u[1] = *(const u32x4*)(Ab + aOff1 + k0 + 16 + g * 8);
    __builtin_prefetch(Ab + aOff0 + k0 + 64, 0, 3);
    __builtin_prefetch(Ab + aOff1 + k0 + 64, 0, 3);

    const u16* bt = &sBt[cur][0];
#pragma unroll
    for (int j = 0; j < 4; ++j) {
      const u16* bp = bt + (j * 16 + l15) * KSTEP + g * 8;
      FragBF fb;
      fb.u[0] = *(const u32x4*)(bp);
      fb.u[1] = *(const u32x4*)(bp + 16);
      acc[0][j] = __builtin_amdgcn_wmma_f32_16x16x32_bf16(
          false, fa0.v, false, fb.v, (short)0, acc[0][j], false, false);
      acc[1][j] = __builtin_amdgcn_wmma_f32_16x16x32_bf16(
          false, fa1.v, false, fb.v, (short)0, acc[1][j], false, false);
    }
    __syncthreads();                             // done reading before rewrite
  }

  // C layout: vgpr rr holds row (+ rr + 8*g), col = nbase + j*16 + l15
#pragma unroll
  for (int h = 0; h < 2; ++h) {
#pragma unroll
    for (int j = 0; j < 4; ++j) {
      size_t col = (size_t)(nbase + j * 16 + l15);
#pragma unroll
      for (int rr = 0; rr < 8; ++rr) {
        Cb[(size_t)(mbase + h * 16 + rr + g * 8) * Nout + col] = acc[h][j][rr];
      }
    }
  }
}

// =====================================================================
// Kernel 3: row softmax of energy/32, fp32 in -> bf16 attn out.
// =====================================================================
__global__ void __launch_bounds__(256) softmax_rows_kernel(
    const float* __restrict__ E, u16* __restrict__ Attn, float scale)
{
  __shared__ float red[256];
  int row = blockIdx.x;
  int t = threadIdx.x;
  f32x4 ev = *(const f32x4*)(E + (size_t)row * 1024 + t * 4);
  ev *= scale;

  float m = fmaxf(fmaxf(ev.x, ev.y), fmaxf(ev.z, ev.w));
  red[t] = m; __syncthreads();
  for (int s = 128; s > 0; s >>= 1) {
    if (t < s) red[t] = fmaxf(red[t], red[t + s]);
    __syncthreads();
  }
  float rmax = red[0];
  __syncthreads();

  float p0 = __expf(ev.x - rmax), p1 = __expf(ev.y - rmax);
  float p2 = __expf(ev.z - rmax), p3 = __expf(ev.w - rmax);
  red[t] = p0 + p1 + p2 + p3; __syncthreads();
  for (int s = 128; s > 0; s >>= 1) {
    if (t < s) red[t] += red[t + s];
    __syncthreads();
  }
  float inv = 1.0f / red[0];

  u32x2 pk;
  pk.x = (unsigned)f2bf(p0 * inv) | ((unsigned)f2bf(p1 * inv) << 16);
  pk.y = (unsigned)f2bf(p2 * inv) | ((unsigned)f2bf(p3 * inv) << 16);
  *(u32x2*)(Attn + (size_t)row * 1024 + t * 4) = pk;
}

// =====================================================================
// Kernel 4: output 1x1 conv + gamma*out + x residual (fp32).
//   outT [B][1024][512] row-major == reference's raw [B][8][256][256] view.
// =====================================================================
__global__ void __launch_bounds__(256) outproj_kernel(
    const float* __restrict__ outT, const float* __restrict__ Wo,
    const float* __restrict__ bo,  const float* __restrict__ gamma,
    const float* __restrict__ x,   float* __restrict__ y)
{
  __shared__ float sWo[512];
  __shared__ float sbo[64];
  int t = threadIdx.x;
  for (int i = t; i < 512; i += 256) sWo[i] = Wo[i];
  if (t < 64) sbo[t] = bo[t];
  __syncthreads();

  float gm = gamma[0];
  int gid = blockIdx.x * 256 + t;
  int b   = gid >> 16;
  int pix = gid & (WH - 1);

  const float* op = outT + (size_t)b * 8 * WH + pix;
  float ov[8];
#pragma unroll
  for (int cq = 0; cq < 8; ++cq) ov[cq] = op[(size_t)cq * WH];

  const float* xp = x + (size_t)b * CCH * WH + pix;
  float*       yp = y + (size_t)b * CCH * WH + pix;
  for (int o = 0; o < 64; ++o) {
    float acc = sbo[o];
#pragma unroll
    for (int cq = 0; cq < 8; ++cq) acc = fmaf(sWo[o * 8 + cq], ov[cq], acc);
    yp[(size_t)o * WH] = fmaf(gm, acc, xp[(size_t)o * WH]);
  }
}

// =====================================================================
extern "C" void kernel_launch(void* const* d_in, const int* in_sizes, int n_in,
                              void* d_out, int out_size, void* d_ws, size_t ws_size,
                              hipStream_t stream)
{
  const float* x     = (const float*)d_in[0];
  const float* Wq    = (const float*)d_in[1];
  const float* bq    = (const float*)d_in[2];
  const float* Wk    = (const float*)d_in[3];
  const float* bk    = (const float*)d_in[4];
  const float* Wv    = (const float*)d_in[5];
  const float* bv    = (const float*)d_in[6];
  const float* Wo    = (const float*)d_in[7];
  const float* bo    = (const float*)d_in[8];
  const float* gamma = (const float*)d_in[9];
  float* y = (float*)d_out;

  char* ws = (char*)d_ws;
  u16*   qT     = (u16*)(ws);                    //  8 MiB  [B][1024][512] bf16
  u16*   kT     = (u16*)(ws + (8ull  << 20));    //  8 MiB  [B][1024][512] bf16
  u16*   vT     = (u16*)(ws + (16ull << 20));    //  8 MiB  [B][512][1024] bf16
  float* energy = (float*)(ws + (24ull << 20));  // 32 MiB  [B][1024][1024] f32
  u16*   attn   = (u16*)(ws + (56ull << 20));    // 16 MiB  [B][1024][1024] bf16
  float* outT   = (float*)(ws + (72ull << 20));  // 16 MiB  [B][1024][512] f32

  dim3 blk(256);
  dim3 gblk(32, 8);   // 8 wave32 waves per block

  // 1) fused q/k/v projection + patch tokenization (fp32 -> bf16)
  qkv_patch_kernel<<<(BATCH * WH) / 256, blk, 0, stream>>>(
      x, Wq, bq, Wk, bk, Wv, bv, qT, kT, vT);

  // 2) energy[b,i,j] = sum_m qT[i,m]*kT[j,m]   (1024x1024x512 per batch)
  //    blocks = 8 * (1024/256) * (1024/64) = 512
  gemm_nt_bf16_wmma<<<512, gblk, 0, stream>>>(
      qT, kT, energy, /*K=*/512, /*Nout=*/1024, /*rowBlocks=*/4, /*nStrips=*/16,
      1024ull * 512, 1024ull * 512, 1024ull * 1024);

  // 3) attn = softmax(energy / sqrt(1024))  (8192 rows)
  softmax_rows_kernel<<<BATCH * NTOK, blk, 0, stream>>>(energy, attn, 0.03125f);

  // 4) outT[b,j,m] = sum_n attn[j,n]*v[m,n]  (1024x512x1024 per batch)
  //    blocks = 8 * (1024/256) * (512/64) = 256
  gemm_nt_bf16_wmma<<<256, gblk, 0, stream>>>(
      attn, vT, outT, /*K=*/1024, /*Nout=*/512, /*rowBlocks=*/4, /*nStrips=*/8,
      1024ull * 1024, 512ull * 1024, 1024ull * 512);

  // 5) y = gamma * conv1x1(outT, Wo, bo) + x
  outproj_kernel<<<(BATCH * WH) / 256, blk, 0, stream>>>(
      outT, Wo, bo, gamma, x, y);
}
